// GATWithJK_12773232738309
// MI455X (gfx1250) — matrix operand order repaired and checked by reading, not compiled, not run on hardware
//
#include <hip/hip_runtime.h>
#include <math.h>

#define NNODES 10000
#define NEDGES 160000
#define NGRAPH 64
#define CDIM   512
#define HEADS  4
#define HIDD   128
#define HLD    768
#define GATE4  3072
#define OUTD   8
#define NEGSL  0.2f
#define LCHUNK 2048

typedef _Float16 v8h  __attribute__((ext_vector_type(8)));
typedef _Float16 v16h __attribute__((ext_vector_type(16)));
typedef float    v8f  __attribute__((ext_vector_type(8)));

__device__ __forceinline__ float sigf(float x){ return 1.0f/(1.0f+expf(-x)); }
__device__ __forceinline__ unsigned fenc(float f){
    unsigned u=__float_as_uint(f); return (u&0x80000000u)? ~u : (u|0x80000000u);
}
__device__ __forceinline__ float fdec(unsigned u){
    unsigned v=(u&0x80000000u)? (u&0x7FFFFFFFu) : ~u; return __uint_as_float(v);
}

// ---------------- WMMA GEMM:  D[Mv x (grid.x*256)] = act(A * B^T + bias) ----------------
// A is f32, rows = Mv, K = K1+K2 split across two sources (A2 may be null, K2=0).
// B is f16, row m = output column m, K-contiguous, leading dim ldb.
// B tile staging uses CDNA5 async memory->LDS copies (ASYNCcnt), bypassing VGPRs.
__global__ __launch_bounds__(256)
void k_gemm(const float* __restrict__ A1, int lda1, int K1,
            const float* __restrict__ A2, int lda2, int K2,
            const _Float16* __restrict__ B, int ldb,
            const float* __restrict__ bias,
            float* __restrict__ D, int ldd, int Mv, int act)
{
    __shared__ __align__(16) _Float16 Al[64*32];
    __shared__ __align__(16) _Float16 Bl[256*32];
    const int t = threadIdx.x;
    const int colBase = blockIdx.x * 256;
    const int rowBase = blockIdx.y * 64;
    const int K = K1 + K2;
    const int lane = t & 31;
    const int wid  = t >> 5;
    const int rbase = (wid >> 2) * 32;   // 2 row-waves
    const int cbase = (wid & 3) * 64;    // 4 col-waves
    const int lh = lane & 15;
    const int hi = lane >> 4;

    v8f acc[2][4];
    for (int i=0;i<2;i++) for (int j=0;j<4;j++) for (int e=0;e<8;e++) acc[i][j][e]=0.0f;

    const int arow = t >> 2;
    const int akq  = (t & 3) * 8;
    // per-thread LDS byte offset for the async B copy (low 32 bits of the
    // generic pointer = LDS-relative byte address)
    const unsigned ldsOff = (unsigned)(size_t)(&Bl[t*32]);

    for (int kt = 0; kt < K; kt += 32) {
        // stage B tile (256 cols x 32 K halves): async global -> LDS, 64B/lane
        {
            unsigned long long ga =
                (unsigned long long)(B + (size_t)(colBase + t)*ldb + kt);
            asm volatile(
                "global_load_async_to_lds_b128 %0, %1, off\n\t"
                "global_load_async_to_lds_b128 %0, %1, off offset:16\n\t"
                "global_load_async_to_lds_b128 %0, %1, off offset:32\n\t"
                "global_load_async_to_lds_b128 %0, %1, off offset:48"
                :: "v"(ldsOff), "v"(ga) : "memory");
        }
        // stage A tile (64 x 32), f32 -> f16 (conversion needs the VGPR path)
        {
            int rg = rowBase + arow;
            float4 f0, f1;
            if (rg < Mv) {
                int kg = kt + akq;
                const float* src = (kg < K1) ? (A1 + (size_t)rg*lda1 + kg)
                                             : (A2 + (size_t)rg*lda2 + (kg - K1));
                f0 = *(const float4*)src;
                f1 = *(const float4*)(src + 4);
            } else {
                f0 = make_float4(0.f,0.f,0.f,0.f); f1 = f0;
            }
            v8h hv;
            hv[0]=(_Float16)f0.x; hv[1]=(_Float16)f0.y; hv[2]=(_Float16)f0.z; hv[3]=(_Float16)f0.w;
            hv[4]=(_Float16)f1.x; hv[5]=(_Float16)f1.y; hv[6]=(_Float16)f1.z; hv[7]=(_Float16)f1.w;
            *(v8h*)&Al[arow*32 + akq] = hv;
        }
        // wait for this wave's async LDS writes, then sync the workgroup
        asm volatile("s_wait_asynccnt 0x0" ::: "memory");
        __syncthreads();

        v16h af[2];
        for (int fr=0; fr<2; fr++){
            int r  = rbase + fr*16 + lh;
            int ko = hi ? 8 : 0;
            v8h lo  = *(const v8h*)&Al[r*32 + ko];
            v8h hi8 = *(const v8h*)&Al[r*32 + 16 + ko];
            af[fr] = __builtin_shufflevector(lo, hi8, 0,1,2,3,4,5,6,7,8,9,10,11,12,13,14,15);
        }
        for (int fc=0; fc<4; fc++){
            int c  = cbase + fc*16 + lh;
            int kb = hi ? 8 : 0;
            v8h lo  = *(const v8h*)&Bl[c*32 + kb];
            v8h hi8 = *(const v8h*)&Bl[c*32 + kb + 16];
            v16h bf = __builtin_shufflevector(lo, hi8, 0,1,2,3,4,5,6,7,8,9,10,11,12,13,14,15);
            for (int fr=0; fr<2; fr++){
                acc[fr][fc] = __builtin_amdgcn_wmma_f32_16x16x32_f16(
                    false, af[fr], false, bf, (short)0, acc[fr][fc], false, false);
            }
        }
        __syncthreads();
    }

    for (int fr=0; fr<2; fr++){
        for (int fc=0; fc<4; fc++){
            int colg = colBase + cbase + fc*16 + lh;
            float bv = bias ? bias[colg] : 0.0f;
            for (int e=0; e<8; e++){
                int rowg = rowBase + rbase + fr*16 + e + 8*hi;
                if (rowg < Mv){
                    float v = acc[fr][fc][e] + bv;
                    if (act == 1) v = fmaxf(v, 0.0f);
                    D[(size_t)rowg*ldd + colg] = v;
                }
            }
        }
    }
}

// ---------------- utility kernels ----------------
__global__ void k_convert(const float* __restrict__ src, _Float16* __restrict__ dst,
                          int M, int K, int ldd, int k0, int trans)
{
    long tot = (long)M * K;
    for (long i = (long)blockIdx.x*blockDim.x + threadIdx.x; i < tot;
         i += (long)gridDim.x*blockDim.x){
        int m = (int)(i / K), k = (int)(i % K);
        float v = trans ? src[(size_t)k*M + m] : src[(size_t)m*K + k];
        dst[(size_t)m*ldd + k0 + k] = (_Float16)v;
    }
}

__global__ void k_fillf(float* p, long n, float v){
    for (long i = (long)blockIdx.x*blockDim.x + threadIdx.x; i < n;
         i += (long)gridDim.x*blockDim.x) p[i] = v;
}
__global__ void k_fill_scalar(float* p, long n, const float* s){
    float v = *s;
    for (long i = (long)blockIdx.x*blockDim.x + threadIdx.x; i < n;
         i += (long)gridDim.x*blockDim.x) p[i] = v;
}
__global__ void k_addvec(const float* a, const float* b, float* o, int n){
    for (int i = blockIdx.x*blockDim.x + threadIdx.x; i < n;
         i += gridDim.x*blockDim.x) o[i] = a[i] + b[i];
}

// ---------------- GAT attention ----------------
__global__ void k_gat_scores(const float* __restrict__ hg, const float* __restrict__ asrc,
                             const float* __restrict__ adst,
                             float* __restrict__ as_, float* __restrict__ ad_)
{
    int tid = blockIdx.x*blockDim.x + threadIdx.x;
    if (tid >= NNODES*HEADS) return;
    int n = tid >> 2, hd = tid & 3;
    const float* hp = hg + (size_t)n*CDIM + hd*HIDD;
    const float* ap = asrc + hd*HIDD;
    const float* dp = adst + hd*HIDD;
    float s = 0.f, d = 0.f;
    for (int k=0;k<HIDD;k++){ float h = hp[k]; s += h*ap[k]; d += h*dp[k]; }
    as_[tid] = s; ad_[tid] = d;
}

__device__ __forceinline__ void edge_sd(const int* ei, long e, int& s, int& d){
    if (e < NEDGES){ s = ei[e]; d = ei[NEDGES + e]; }
    else           { s = d = (int)(e - NEDGES); }
}

__global__ void k_edge_max(const int* __restrict__ ei, const float* __restrict__ as_,
                           const float* __restrict__ ad_, unsigned* __restrict__ segm, long tot)
{
    for (long tid = (long)blockIdx.x*blockDim.x + threadIdx.x; tid < tot;
         tid += (long)gridDim.x*blockDim.x){
        long e = tid >> 2; int hd = (int)(tid & 3), s, d;
        edge_sd(ei, e, s, d);
        float v = as_[s*4+hd] + ad_[d*4+hd];
        v = v > 0.f ? v : NEGSL*v;
        atomicMax(&segm[d*4+hd], fenc(v));
    }
}
__global__ void k_edge_den(const int* __restrict__ ei, const float* __restrict__ as_,
                           const float* __restrict__ ad_, const unsigned* __restrict__ segm,
                           float* __restrict__ den, float* __restrict__ ab, long tot)
{
    for (long tid = (long)blockIdx.x*blockDim.x + threadIdx.x; tid < tot;
         tid += (long)gridDim.x*blockDim.x){
        long e = tid >> 2; int hd = (int)(tid & 3), s, d;
        edge_sd(ei, e, s, d);
        float v = as_[s*4+hd] + ad_[d*4+hd];
        v = v > 0.f ? v : NEGSL*v;
        float ex = expf(v - fdec(segm[d*4+hd]));
        ab[tid] = ex;
        atomicAdd(&den[d*4+hd], ex);
    }
}
__global__ void k_edge_div(const int* __restrict__ ei, const float* __restrict__ den,
                           float* __restrict__ ab, long tot)
{
    for (long tid = (long)blockIdx.x*blockDim.x + threadIdx.x; tid < tot;
         tid += (long)gridDim.x*blockDim.x){
        long e = tid >> 2; int hd = (int)(tid & 3), s, d;
        edge_sd(ei, e, s, d);
        ab[tid] = ab[tid] / den[d*4+hd];
    }
}
__global__ void k_scatter(const int* __restrict__ ei, const float* __restrict__ hg,
                          const float* __restrict__ ab, float* __restrict__ out, long tot)
{
    for (long idx = (long)blockIdx.x*blockDim.x + threadIdx.x; idx < tot;
         idx += (long)gridDim.x*blockDim.x){
        long e = idx >> 9; int c = (int)(idx & 511); int hd = c >> 7, s, d;
        edge_sd(ei, e, s, d);
        float w = ab[e*4 + hd];
        atomicAdd(&out[(size_t)d*CDIM + c], w * hg[(size_t)s*CDIM + c]);
    }
}
__global__ void k_bias_relu(float* __restrict__ x, const float* __restrict__ b, long tot){
    for (long i = (long)blockIdx.x*blockDim.x + threadIdx.x; i < tot;
         i += (long)gridDim.x*blockDim.x){
        int c = (int)(i & 511);
        x[i] = fmaxf(x[i] + b[c], 0.0f);
    }
}

// ---------------- LSTM cell + fused JK attention dot ----------------
__global__ __launch_bounds__(256)
void k_lstm_cell(const float* __restrict__ gates, float* __restrict__ h, float* __restrict__ c,
                 const float* __restrict__ attw, float* __restrict__ scoreRow, int r0)
{
    __shared__ float red[256];
    int n = r0 + blockIdx.x;
    const float* g = gates + (size_t)blockIdx.x * GATE4;
    int t = threadIdx.x;
    float local = 0.f;
    for (int ii=0; ii<3; ii++){
        int j = ii*256 + t;
        float gi = g[j], gf = g[HLD + j], gg = g[2*HLD + j], go = g[3*HLD + j];
        size_t off = (size_t)n*HLD + j;
        float nc = sigf(gf)*c[off] + sigf(gi)*tanhf(gg);
        float nh = sigf(go)*tanhf(nc);
        c[off] = nc; h[off] = nh;
        local += nh * attw[j];
    }
    red[t] = local; __syncthreads();
    for (int s=128; s>0; s>>=1){ if (t < s) red[t] += red[t+s]; __syncthreads(); }
    if (t == 0) scoreRow[n] += red[0];
}

// ---------------- JK softmax + mean pool ----------------
__global__ __launch_bounds__(256)
void k_jk_pool(const float* __restrict__ score, const float* __restrict__ X0,
               const float* __restrict__ X1, const float* __restrict__ X2,
               const int* __restrict__ batch, float* __restrict__ pool)
{
    int n = blockIdx.x;
    float s0 = score[n], s1 = score[NNODES+n], s2 = score[2*NNODES+n];
    float m = fmaxf(s0, fmaxf(s1, s2));
    float e0 = expf(s0-m), e1 = expf(s1-m), e2 = expf(s2-m);
    float inv = 1.0f/(e0+e1+e2);
    float a0 = e0*inv, a1 = e1*inv, a2 = e2*inv;
    int g = batch[n];
    for (int cidx = threadIdx.x; cidx < CDIM; cidx += 256){
        size_t off = (size_t)n*CDIM + cidx;
        float v = a0*X0[off] + a1*X1[off] + a2*X2[off];
        atomicAdd(&pool[(size_t)g*CDIM + cidx], v);
    }
}
__global__ void k_count(const int* __restrict__ batch, float* __restrict__ cnt){
    for (int n = blockIdx.x*blockDim.x + threadIdx.x; n < NNODES; n += gridDim.x*blockDim.x)
        atomicAdd(&cnt[batch[n]], 1.0f);
}
__global__ void k_pool_div(const float* __restrict__ pool, const float* __restrict__ cnt,
                           float* __restrict__ g){
    int tid = blockIdx.x*blockDim.x + threadIdx.x;
    if (tid >= NGRAPH*CDIM) return;
    g[tid] = pool[tid] / fmaxf(cnt[tid >> 9], 1.0f);
}
__global__ void k_fc3(const float* __restrict__ g2, const float* __restrict__ W,
                      const float* __restrict__ b, float* __restrict__ out){
    int tid = blockIdx.x*blockDim.x + threadIdx.x;
    if (tid >= NGRAPH*OUTD) return;
    int gi = tid >> 3, o = tid & 7;
    float s = b[o];
    for (int k=0;k<CDIM;k++) s += g2[(size_t)gi*CDIM + k] * W[(size_t)k*OUTD + o];
    out[tid] = s;
}

// ---------------- host orchestration ----------------
extern "C" void kernel_launch(void* const* d_in, const int* in_sizes, int n_in,
                              void* d_out, int out_size, void* d_ws, size_t ws_size,
                              hipStream_t stream)
{
    (void)in_sizes; (void)n_in; (void)out_size;
    const float* x    = (const float*)d_in[0];
    const int*   ei   = (const int*)  d_in[1];
    const int*   batch= (const int*)  d_in[2];
    const float* Wg[3]   = {(const float*)d_in[3],  (const float*)d_in[7],  (const float*)d_in[11]};
    const float* asrc[3] = {(const float*)d_in[4],  (const float*)d_in[8],  (const float*)d_in[12]};
    const float* adst[3] = {(const float*)d_in[5],  (const float*)d_in[9],  (const float*)d_in[13]};
    const float* bgat[3] = {(const float*)d_in[6],  (const float*)d_in[10], (const float*)d_in[14]};
    const float* Wih[2]  = {(const float*)d_in[15], (const float*)d_in[19]};
    const float* Whh[2]  = {(const float*)d_in[16], (const float*)d_in[20]};
    const float* bih[2]  = {(const float*)d_in[17], (const float*)d_in[21]};
    const float* bhh[2]  = {(const float*)d_in[18], (const float*)d_in[22]};
    const float* attw = (const float*)d_in[23];
    const float* attb = (const float*)d_in[24];
    const float* fc1W = (const float*)d_in[25]; const float* fc1b = (const float*)d_in[26];
    const float* fc2W = (const float*)d_in[27]; const float* fc2b = (const float*)d_in[28];
    const float* fc3W = (const float*)d_in[29]; const float* fc3b = (const float*)d_in[30];
    float* out = (float*)d_out;

    char* base = (char*)d_ws; size_t off = 0;
    auto carve = [&](size_t bytes)->void*{
        void* p = base + off; off = (off + bytes + 255) & ~(size_t)255; return p;
    };
    float*    hg    = (float*)   carve((size_t)NNODES*CDIM*4);
    float*    X[3]  = {(float*)carve((size_t)NNODES*CDIM*4),
                       (float*)carve((size_t)NNODES*CDIM*4),
                       (float*)carve((size_t)NNODES*CDIM*4)};
    _Float16* Wt[3] = {(_Float16*)carve((size_t)CDIM*128*2),
                       (_Float16*)carve((size_t)CDIM*CDIM*2),
                       (_Float16*)carve((size_t)CDIM*CDIM*2)};
    _Float16* wcat[2] = {(_Float16*)carve((size_t)GATE4*1280*2),
                         (_Float16*)carve((size_t)GATE4*1280*2)};
    _Float16* fc1t = (_Float16*)carve((size_t)CDIM*CDIM*2);
    _Float16* fc2t = (_Float16*)carve((size_t)CDIM*CDIM*2);
    float* bsum[2] = {(float*)carve(GATE4*4), (float*)carve(GATE4*4)};
    float*    as_  = (float*)   carve((size_t)NNODES*HEADS*4);
    float*    ad_  = (float*)   carve((size_t)NNODES*HEADS*4);
    unsigned* segm = (unsigned*)carve((size_t)NNODES*HEADS*4);
    float*    den  = (float*)   carve((size_t)NNODES*HEADS*4);
    long Etot = (long)NEDGES + NNODES;
    float*    ab   = (float*)   carve((size_t)Etot*HEADS*4);
    float*    hbuf = (float*)   carve((size_t)NNODES*HLD*4);
    float*    cbuf = (float*)   carve((size_t)NNODES*HLD*4);
    float*    gates= (float*)   carve((size_t)LCHUNK*GATE4*4);
    float*    score= (float*)   carve((size_t)3*NNODES*4);
    float*    pool = (float*)   carve((size_t)NGRAPH*CDIM*4);
    float*    cnt  = (float*)   carve((size_t)NGRAPH*4);
    float*    gvec = (float*)   carve((size_t)NGRAPH*CDIM*4);
    float*    g1   = (float*)   carve((size_t)NGRAPH*CDIM*4);
    float*    g2   = (float*)   carve((size_t)NGRAPH*CDIM*4);
    if (off > ws_size) return;  // workspace too small; deterministic no-op

    const int TB = 256;
    auto gsz = [](long n, int tb){ long b = (n + tb - 1)/tb; return (int)(b > 4096 ? 4096 : b); };

    // ---- weight conversions (f32 -> f16, K-contiguous output-column-major) ----
    k_convert<<<gsz((long)CDIM*128,TB),TB,0,stream>>>(Wg[0], Wt[0], CDIM, 128, 128, 0, 1);
    k_convert<<<gsz((long)CDIM*CDIM,TB),TB,0,stream>>>(Wg[1], Wt[1], CDIM, CDIM, CDIM, 0, 1);
    k_convert<<<gsz((long)CDIM*CDIM,TB),TB,0,stream>>>(Wg[2], Wt[2], CDIM, CDIM, CDIM, 0, 1);
    for (int d=0; d<2; d++){
        k_convert<<<gsz((long)GATE4*CDIM,TB),TB,0,stream>>>(Wih[d], wcat[d], GATE4, CDIM, 1280, 0, 0);
        k_convert<<<gsz((long)GATE4*HLD,TB),TB,0,stream>>>(Whh[d], wcat[d], GATE4, HLD, 1280, CDIM, 0);
        k_addvec<<<gsz(GATE4,TB),TB,0,stream>>>(bih[d], bhh[d], bsum[d], GATE4);
    }
    k_convert<<<gsz((long)CDIM*CDIM,TB),TB,0,stream>>>(fc1W, fc1t, CDIM, CDIM, CDIM, 0, 1);
    k_convert<<<gsz((long)CDIM*CDIM,TB),TB,0,stream>>>(fc2W, fc2t, CDIM, CDIM, CDIM, 0, 1);

    // ---- GAT layers ----
    const float* inAct = x; int Kin = 128;
    for (int l=0; l<3; l++){
        dim3 gg(CDIM/256, (NNODES + 63)/64);
        k_gemm<<<gg, TB, 0, stream>>>(inAct, Kin, Kin, (const float*)nullptr, 0, 0,
                                      Wt[l], Kin, (const float*)nullptr,
                                      hg, CDIM, NNODES, 0);
        k_gat_scores<<<(NNODES*HEADS + TB-1)/TB, TB, 0, stream>>>(hg, asrc[l], adst[l], as_, ad_);
        k_fillf<<<gsz((long)NNODES*HEADS,TB),TB,0,stream>>>((float*)segm, (long)NNODES*HEADS, 0.0f);
        k_fillf<<<gsz((long)NNODES*HEADS,TB),TB,0,stream>>>(den, (long)NNODES*HEADS, 0.0f);
        k_fillf<<<gsz((long)NNODES*CDIM,TB),TB,0,stream>>>(X[l], (long)NNODES*CDIM, 0.0f);
        long te = Etot*HEADS;
        k_edge_max<<<gsz(te,TB),TB,0,stream>>>(ei, as_, ad_, segm, te);
        k_edge_den<<<gsz(te,TB),TB,0,stream>>>(ei, as_, ad_, segm, den, ab, te);
        k_edge_div<<<gsz(te,TB),TB,0,stream>>>(ei, den, ab, te);
        long ts = Etot*(long)CDIM;
        k_scatter<<<gsz(ts,TB),TB,0,stream>>>(ei, hg, ab, X[l], ts);
        k_bias_relu<<<gsz((long)NNODES*CDIM,TB),TB,0,stream>>>(X[l], bgat[l], (long)NNODES*CDIM);
        inAct = X[l]; Kin = CDIM;
    }

    // ---- bidirectional JK-LSTM with fused attention score ----
    k_fill_scalar<<<gsz((long)3*NNODES,TB),TB,0,stream>>>(score, (long)3*NNODES, attb);
    for (int dir=0; dir<2; dir++){
        k_fillf<<<gsz((long)NNODES*HLD,TB),TB,0,stream>>>(hbuf, (long)NNODES*HLD, 0.0f);
        k_fillf<<<gsz((long)NNODES*HLD,TB),TB,0,stream>>>(cbuf, (long)NNODES*HLD, 0.0f);
        for (int s=0; s<3; s++){
            int l = (dir == 0) ? s : (2 - s);
            for (int r0 = 0; r0 < NNODES; r0 += LCHUNK){
                int Mv = NNODES - r0; if (Mv > LCHUNK) Mv = LCHUNK;
                dim3 gg(GATE4/256, (Mv + 63)/64);
                k_gemm<<<gg, TB, 0, stream>>>(X[l] + (size_t)r0*CDIM, CDIM, CDIM,
                                              hbuf + (size_t)r0*HLD, HLD, HLD,
                                              wcat[dir], 1280, bsum[dir],
                                              gates, GATE4, Mv, 0);
                k_lstm_cell<<<Mv, TB, 0, stream>>>(gates, hbuf, cbuf,
                                                   attw + dir*HLD, score + (size_t)l*NNODES, r0);
            }
        }
    }

    // ---- JK softmax, mean pool, MLP head ----
    k_fillf<<<gsz((long)NGRAPH*CDIM,TB),TB,0,stream>>>(pool, (long)NGRAPH*CDIM, 0.0f);
    k_fillf<<<gsz(NGRAPH,TB),TB,0,stream>>>(cnt, NGRAPH, 0.0f);
    k_jk_pool<<<NNODES, TB, 0, stream>>>(score, X[0], X[1], X[2], batch, pool);
    k_count<<<gsz(NNODES,TB),TB,0,stream>>>(batch, cnt);
    k_pool_div<<<(NGRAPH*CDIM + TB-1)/TB, TB, 0, stream>>>(pool, cnt, gvec);

    dim3 gf(CDIM/256, 1);
    k_gemm<<<gf, TB, 0, stream>>>(gvec, CDIM, CDIM, (const float*)nullptr, 0, 0,
                                  fc1t, CDIM, fc1b, g1, CDIM, NGRAPH, 1);
    k_gemm<<<gf, TB, 0, stream>>>(g1, CDIM, CDIM, (const float*)nullptr, 0, 0,
                                  fc2t, CDIM, fc2b, g2, CDIM, NGRAPH, 1);
    k_fc3<<<(NGRAPH*OUTD + TB-1)/TB, TB, 0, stream>>>(g2, fc3W, fc3b, out);
}
